// SelfAttention_49211735277774
// MI455X (gfx1250) — matrix operand order repaired and checked by reading, not compiled, not run on hardware
//
#include <hip/hip_runtime.h>
#include <hip/hip_bf16.h>

// ---------------------------------------------------------------------------
// SAGAN-style self attention, B=8 C=256 N=4096 d=32, fused flash-style.
// softmax is over the *query* axis i:  attn[i,j] = softmax_i(q_i . k_j)
// ctx[c,j] = sum_i v[c,i] * attn[i,j]  ->  standard attention with q/k swapped.
//
// Attention kernel: block = 8 waves = (2 channel halves) x (4 j-groups of 32
// columns) covering 128 columns of one batch. Per 32-wide i-block the shared
// v tile (256x32 bf16, 16 KB) and q tile (32x32 bf16, 2 KB) are DMA'd into
// LDS by the Tensor Data Mover (double buffered, TENSORcnt-synchronized),
// then each wave runs 4 score WMMAs + online softmax + 16 ctx WMMAs.
// ---------------------------------------------------------------------------

typedef __attribute__((ext_vector_type(16))) __bf16 bf16x16;
typedef __attribute__((ext_vector_type(8)))  __bf16 bf16x8;
typedef __attribute__((ext_vector_type(8)))  float  f32x8;
typedef __attribute__((ext_vector_type(4)))  unsigned int u32x4;
typedef __attribute__((ext_vector_type(8)))  int i32x8;
typedef __attribute__((ext_vector_type(4)))  int i32x4;

#define BATCH 8
#define CH    256
#define NPIX  4096
#define DD    32

// workspace layout (bytes), all 256-aligned
#define OFF_SIG 0u
#define OFF_WQ  256u
#define OFF_WK  (OFF_WQ + 32u*256u*2u)
#define OFF_WV  (OFF_WK + 32u*256u*2u)
#define OFF_QT  (OFF_WV + 256u*256u*2u)               // qT[b][n][32] bf16
#define OFF_KT  (OFF_QT + (size_t)BATCH*NPIX*DD*2u)   // kT[b][n][32] bf16
#define OFF_V   (OFF_KT + (size_t)BATCH*NPIX*DD*2u)   // v[b][c][n]  bf16

__device__ __forceinline__ bf16x16 cat8(bf16x8 a, bf16x8 b) {
  return __builtin_shufflevector(a, b, 0,1,2,3,4,5,6,7,8,9,10,11,12,13,14,15);
}

// low 32 bits of a generic LDS pointer == LDS byte offset (aperture in [63:32])
__device__ __forceinline__ unsigned int lds_off(const void* p) {
  return (unsigned int)(unsigned long long)p;
}

// Tensor Data Mover: 2D tile load, 2-byte elements (ISA 8.3-8.6 D# layout).
// amdgpu-toolchain (clang-23) uses the 6-arg builtin: the 5th operand is the
// unused VADDR4 slot of the VIMAGE encoding -> zero-filled.
__device__ __forceinline__ void tdm_load_2d(unsigned int ldsoff, unsigned long long gaddr,
                                            unsigned int tile_x, unsigned int tile_y,
                                            unsigned int tensor_x, unsigned int tensor_y,
                                            unsigned long long stride0_elems) {
  u32x4 g0;
  g0[0] = 1u;                                   // count=1, user descriptor
  g0[1] = ldsoff;                               // lds_addr (bytes)
  g0[2] = (unsigned int)gaddr;                  // global_addr[31:0]
  g0[3] = ((unsigned int)(gaddr >> 32) & 0x01FFFFFFu) | (2u << 30); // addr[56:32], type=2
  i32x8 g1;
  g1[0] = (int)(1u << 16);                      // wg_mask=0, data_size=1 (2B), no iter/pad
  g1[1] = (int)((tensor_x & 0xFFFFu) << 16);    // tensor_dim0[15:0] @ bits63:48
  g1[2] = (int)((tensor_x >> 16) | ((tensor_y & 0xFFFFu) << 16)); // dim0 hi | dim1 lo
  g1[3] = (int)((tensor_y >> 16) | (tile_x << 16));               // dim1 hi | tile_dim0
  g1[4] = (int)tile_y;                          // tile_dim1 (tile_dim2=0 -> unused)
  g1[5] = (int)(unsigned int)stride0_elems;     // tensor_dim0_stride[31:0]
  g1[6] = (int)((unsigned int)(stride0_elems >> 32) & 0xFFFFu);   // stride0[47:32]
  g1[7] = 0;
  i32x4 g2; g2[0] = 1; g2[1] = 1; g2[2] = 0; g2[3] = 0;  // tensor_dim2/3=1, tile_dim3=0
  i32x4 g3; g3[0] = 0; g3[1] = (int)(1u << 16); g3[2] = 0; g3[3] = 0; // tensor_dim4=1
  i32x8 g4 = {0, 0, 0, 0, 0, 0, 0, 0};          // unused VADDR4 slot
  __builtin_amdgcn_tensor_load_to_lds(g0, g1, g2, g3, g4, 0);
}

// ---------------------------------------------------------------------------
// Kernel 1: spectral norms (converged power iteration) + bf16 weight prep.
// ---------------------------------------------------------------------------
__global__ __launch_bounds__(256) void sigma_convert_kernel(
    const float* __restrict__ Wq, const float* __restrict__ Wk,
    const float* __restrict__ Wv, char* __restrict__ ws) {
  __shared__ float u[256];
  __shared__ float t[256];
  __shared__ float red[256];
  const int tid = threadIdx.x;
  float invs[3];
  const float* Ws[3] = {Wq, Wk, Wv};
  const int    Rs[3] = {32, 32, 256};
  for (int m = 0; m < 3; ++m) {
    const float* W = Ws[m];
    const int R = Rs[m];
    if (tid < R) u[tid] = 1.0f + 0.013f * (float)tid;
    __syncthreads();
    for (int it = 0; it < 64; ++it) {
      float a = 0.f;                        // t = W^T u
      for (int d = 0; d < R; ++d) a += W[d * 256 + tid] * u[d];
      t[tid] = a;
      __syncthreads();
      float a2 = 0.f;                       // u' = W t
      if (tid < R)
        for (int c = 0; c < 256; ++c) a2 += W[tid * 256 + c] * t[c];
      red[tid] = (tid < R) ? a2 * a2 : 0.f;
      __syncthreads();
      for (int s = 128; s > 0; s >>= 1) {
        if (tid < s) red[tid] += red[tid + s];
        __syncthreads();
      }
      const float rn = rsqrtf(red[0] + 1e-30f);
      __syncthreads();
      if (tid < R) u[tid] = a2 * rn;
      __syncthreads();
    }
    float a = 0.f;                          // sigma = ||W^T u||
    for (int d = 0; d < R; ++d) a += W[d * 256 + tid] * u[d];
    red[tid] = a * a;
    __syncthreads();
    for (int s = 128; s > 0; s >>= 1) {
      if (tid < s) red[tid] += red[tid + s];
      __syncthreads();
    }
    invs[m] = rsqrtf(red[0] + 1e-30f);
    __syncthreads();
  }
  if (tid < 3) ((float*)(ws + OFF_SIG))[tid] = invs[tid];
  __bf16* wq = (__bf16*)(ws + OFF_WQ);
  __bf16* wk = (__bf16*)(ws + OFF_WK);
  __bf16* wv = (__bf16*)(ws + OFF_WV);
  for (int i = tid; i < 32 * 256; i += 256)  wq[i] = (__bf16)(Wq[i] * invs[0]);
  for (int i = tid; i < 32 * 256; i += 256)  wk[i] = (__bf16)(Wk[i] * invs[1]);
  for (int i = tid; i < 256 * 256; i += 256) wv[i] = (__bf16)(Wv[i] * invs[2]);
}

// ---------------------------------------------------------------------------
// Kernel 2: q/k/v projections via WMMA.
// ---------------------------------------------------------------------------
__global__ __launch_bounds__(128) void proj_kernel(
    const float* __restrict__ x, const float* __restrict__ bq,
    const float* __restrict__ bk, const float* __restrict__ bv,
    char* __restrict__ ws) {
  __shared__ __align__(64) __bf16 xT[64][256];   // 32 KB
  const int b  = blockIdx.y;
  const int n0 = blockIdx.x * 64;
  const int tid = threadIdx.x;
  {
    const int n = tid & 63, rr = tid >> 6;
    for (int r = rr; r < 256; r += 2)
      xT[n][r] = (__bf16)x[(((size_t)(b * 256 + r)) << 12) + n0 + n];
  }
  __syncthreads();

  const int lane = tid & 31;
  const int wv_id = tid >> 5;
  const int jl = lane & 15;
  const int hh = lane >> 4;
  const int nloc = wv_id * 16 + jl;

  const __bf16* WqB = (const __bf16*)(ws + OFF_WQ);
  const __bf16* WkB = (const __bf16*)(ws + OFF_WK);
  const __bf16* WvB = (const __bf16*)(ws + OFF_WV);
  __bf16* qT = (__bf16*)(ws + OFF_QT);
  __bf16* kT = (__bf16*)(ws + OFF_KT);
  __bf16* vB = (__bf16*)(ws + OFF_V);

  for (int ot = 0; ot < 20; ++ot) {
    const __bf16* Wb; const float* bias; int rbase;
    if (ot < 2)      { Wb = WqB; bias = bq; rbase = ot * 16; }
    else if (ot < 4) { Wb = WkB; bias = bk; rbase = (ot - 2) * 16; }
    else             { Wb = WvB; bias = bv; rbase = (ot - 4) * 16; }
    f32x8 acc = {0.f,0.f,0.f,0.f,0.f,0.f,0.f,0.f};
    const __bf16* wrow = Wb + (size_t)(rbase + jl) * 256;
#pragma unroll
    for (int kc = 0; kc < 8; ++kc) {
      bf16x16 A = cat8(*(const bf16x8*)(wrow + kc * 32 + hh * 16),
                       *(const bf16x8*)(wrow + kc * 32 + hh * 16 + 8));
      bf16x16 Bm = *(const bf16x16*)&xT[nloc][kc * 32 + hh * 16];
      acc = __builtin_amdgcn_wmma_f32_16x16x32_bf16(false, A, false, Bm,
                                                    (short)0, acc, false, false);
    }
    const int n = n0 + nloc;
#pragma unroll
    for (int r = 0; r < 8; ++r) {
      const int row = rbase + r + 8 * hh;
      const __bf16 ob = (__bf16)(acc[r] + bias[row]);
      if (ot < 2)      qT[((size_t)b * NPIX + n) * DD + row] = ob;
      else if (ot < 4) kT[((size_t)b * NPIX + n) * DD + row] = ob;
      else             vB[(((size_t)(b * 256 + row)) << 12) + n] = ob;
    }
  }
}

// ---------------------------------------------------------------------------
// Kernel 3: fused flash attention with TDM-staged, double-buffered LDS tiles.
// ---------------------------------------------------------------------------
__device__ __forceinline__ float softmax_update(f32x8& Sa, f32x8& Sb,
                                                float& m, float& l) {
  float mt = Sa[0];
#pragma unroll
  for (int r = 1; r < 8; ++r) mt = fmaxf(mt, Sa[r]);
#pragma unroll
  for (int r = 0; r < 8; ++r) mt = fmaxf(mt, Sb[r]);
  mt = fmaxf(mt, __shfl_xor(mt, 16, 32));   // combine the two i half-rows
  const float mn = fmaxf(m, mt);
  const float sc = __expf(m - mn);
  float es = 0.f;
#pragma unroll
  for (int r = 0; r < 8; ++r) { Sa[r] = __expf(Sa[r] - mn); es += Sa[r]; }
#pragma unroll
  for (int r = 0; r < 8; ++r) { Sb[r] = __expf(Sb[r] - mn); es += Sb[r]; }
  es += __shfl_xor(es, 16, 32);
  l = l * sc + es;
  m = mn;
  return sc;
}

__global__ __launch_bounds__(256) void attn_kernel(
    const float* __restrict__ xin, const float* __restrict__ gptr,
    float* __restrict__ out, char* __restrict__ ws) {
  __shared__ __align__(64) __bf16 Vt[2][256][32];    // 32 KB, v[c][i] tiles
  __shared__ __align__(64) __bf16 Qt[2][32][32];     //  4 KB, q[i][d] tiles
  __shared__ __align__(64) __bf16 Pt[8][2][16][32];  // 16 KB, per-wave P relayout
  const int tid  = threadIdx.x;
  const int lane = tid & 31;
  const int wid  = tid >> 5;
  const int b    = blockIdx.x >> 5;         // 32 blocks per batch
  const int j0   = (blockIdx.x & 31) * 128; // block covers 128 columns
  const int jg   = wid & 3;                 // 32-column group within block
  const int chh  = wid >> 2;                // channel half (128 ch each)
  const int jbase = j0 + jg * 32;
  const int jl = lane & 15;
  const int hh = lane >> 4;

  const __bf16* qT = (const __bf16*)(ws + OFF_QT) + (size_t)b * NPIX * DD;
  const __bf16* kT = (const __bf16*)(ws + OFF_KT) + (size_t)b * NPIX * DD;
  const __bf16* vB = (const __bf16*)(ws + OFF_V)  + ((size_t)b << 20);
  const unsigned long long vbase = (unsigned long long)(size_t)vB;
  const unsigned long long qbase = (unsigned long long)(size_t)qT;

  // resident k fragments (B matrix, K=d=32) for the two 16-col j-subtiles
  bf16x16 Bk0 = *(const bf16x16*)(kT + (size_t)(jbase + jl) * DD + hh * 16);
  bf16x16 Bk1 = *(const bf16x16*)(kT + (size_t)(jbase + 16 + jl) * DD + hh * 16);

  f32x8 acc[8][2];
#pragma unroll
  for (int ct = 0; ct < 8; ++ct)
#pragma unroll
    for (int js = 0; js < 2; ++js)
      acc[ct][js] = (f32x8){0.f,0.f,0.f,0.f,0.f,0.f,0.f,0.f};
  float m0 = -__builtin_inff(), l0 = 0.f;
  float m1 = -__builtin_inff(), l1 = 0.f;
  const f32x8 z8 = {0.f,0.f,0.f,0.f,0.f,0.f,0.f,0.f};

  if (wid == 0) {  // prologue: DMA first i-block tile (EXEC-independent wave op)
    tdm_load_2d(lds_off(&Vt[0][0][0]), vbase, 32, 256, NPIX, 256, NPIX);
    tdm_load_2d(lds_off(&Qt[0][0][0]), qbase, 1024, 1, 1u << 20, 1, 1024);
  }

  for (int it = 0; it < NPIX / 32; ++it) {
    const int buf = it & 1;
    if (wid == 0) __builtin_amdgcn_s_wait_tensorcnt(0);  // tile `it` landed
    __syncthreads();
    if (wid == 0 && it + 1 < NPIX / 32) {   // prefetch next tile, overlaps compute
      const unsigned long long nb = (unsigned long long)(it + 1) * 32u;
      tdm_load_2d(lds_off(&Vt[buf ^ 1][0][0]), vbase + nb * 2u,
                  32, 256, NPIX, 256, NPIX);
      tdm_load_2d(lds_off(&Qt[buf ^ 1][0][0]), qbase + nb * DD * 2u,
                  1024, 1, 1u << 20, 1, 1024);
    }
    // q fragments (A: M=i, K=d) from LDS
    bf16x16 A0 = cat8(*(const bf16x8*)&Qt[buf][jl][hh * 16],
                      *(const bf16x8*)&Qt[buf][jl][hh * 16 + 8]);
    bf16x16 A1 = cat8(*(const bf16x8*)&Qt[buf][16 + jl][hh * 16],
                      *(const bf16x8*)&Qt[buf][16 + jl][hh * 16 + 8]);
    // scores: 2 i-subtiles x 2 j-subtiles, full d=32 in one WMMA each
    f32x8 S00 = __builtin_amdgcn_wmma_f32_16x16x32_bf16(false, A0, false, Bk0,
                                                        (short)0, z8, false, false);
    f32x8 S10 = __builtin_amdgcn_wmma_f32_16x16x32_bf16(false, A1, false, Bk0,
                                                        (short)0, z8, false, false);
    f32x8 S01 = __builtin_amdgcn_wmma_f32_16x16x32_bf16(false, A0, false, Bk1,
                                                        (short)0, z8, false, false);
    f32x8 S11 = __builtin_amdgcn_wmma_f32_16x16x32_bf16(false, A1, false, Bk1,
                                                        (short)0, z8, false, false);
    const float sc0 = softmax_update(S00, S10, m0, l0);
    const float sc1 = softmax_update(S01, S11, m1, l1);
#pragma unroll
    for (int ct = 0; ct < 8; ++ct) { acc[ct][0] *= sc0; acc[ct][1] *= sc1; }
    // P relayout D-format -> B-format via wave-private LDS [j][i]
    bf16x8 P00, P10, P01, P11;
#pragma unroll
    for (int r = 0; r < 8; ++r) {
      P00[r] = (__bf16)S00[r]; P10[r] = (__bf16)S10[r];
      P01[r] = (__bf16)S01[r]; P11[r] = (__bf16)S11[r];
    }
    *(bf16x8*)&Pt[wid][0][jl][hh * 8]      = P00;   // i = 8*hh + r
    *(bf16x8*)&Pt[wid][0][jl][16 + hh * 8] = P10;   // i = 16 + 8*hh + r
    *(bf16x8*)&Pt[wid][1][jl][hh * 8]      = P01;
    *(bf16x8*)&Pt[wid][1][jl][16 + hh * 8] = P11;
    asm volatile("s_wait_dscnt 0x0" ::: "memory");  // intra-wave LDS turnaround
    bf16x16 Bp0 = *(const bf16x16*)&Pt[wid][0][jl][hh * 16];
    bf16x16 Bp1 = *(const bf16x16*)&Pt[wid][1][jl][hh * 16];
    // ctx: 8 channel tiles x 2 j-subtiles; Av fragment reused across j
#pragma unroll
    for (int ct = 0; ct < 8; ++ct) {
      const __bf16* vr = &Vt[buf][chh * 128 + ct * 16 + jl][0];
      bf16x16 Av = cat8(*(const bf16x8*)(vr + hh * 16),
                        *(const bf16x8*)(vr + hh * 16 + 8));
      acc[ct][0] = __builtin_amdgcn_wmma_f32_16x16x32_bf16(false, Av, false, Bp0,
                                                           (short)0, acc[ct][0],
                                                           false, false);
      acc[ct][1] = __builtin_amdgcn_wmma_f32_16x16x32_bf16(false, Av, false, Bp1,
                                                           (short)0, acc[ct][1],
                                                           false, false);
    }
    __syncthreads();   // everyone done reading `buf` before it is re-filled
  }
  // epilogue: normalize, gamma residual
  const float g = *gptr;
  const float il0 = 1.0f / l0, il1 = 1.0f / l1;
#pragma unroll
  for (int ct = 0; ct < 8; ++ct) {
#pragma unroll
    for (int js = 0; js < 2; ++js) {
      const float il = js ? il1 : il0;
      const int n = jbase + js * 16 + jl;
#pragma unroll
      for (int r = 0; r < 8; ++r) {
        const int c = chh * 128 + ct * 16 + r + 8 * hh;
        const size_t idx = (((size_t)(b * 256 + c)) << 12) + n;
        out[idx] = g * (acc[ct][js][r] * il) + xin[idx];
      }
    }
  }
}

// ---------------------------------------------------------------------------
extern "C" void kernel_launch(void* const* d_in, const int* in_sizes, int n_in,
                              void* d_out, int out_size, void* d_ws, size_t ws_size,
                              hipStream_t stream) {
  const float* x     = (const float*)d_in[0];
  const float* Wq    = (const float*)d_in[1];
  const float* bq    = (const float*)d_in[2];
  const float* Wk    = (const float*)d_in[3];
  const float* bk    = (const float*)d_in[4];
  const float* Wv    = (const float*)d_in[5];
  const float* bv    = (const float*)d_in[6];
  const float* gamma = (const float*)d_in[7];
  float* out = (float*)d_out;
  char*  ws  = (char*)d_ws;

  sigma_convert_kernel<<<1, 256, 0, stream>>>(Wq, Wk, Wv, ws);
  proj_kernel<<<dim3(64, 8), 128, 0, stream>>>(x, bq, bk, bv, ws);
  attn_kernel<<<256, 256, 0, stream>>>(x, gamma, out, ws);
}
